// SelfAttn_22050362098063
// MI455X (gfx1250) — matrix operand order repaired and checked by reading, not compiled, not run on hardware
//
#include <hip/hip_runtime.h>

// ---------------------------------------------------------------------------
// Problem constants (match reference): B=4, S=1024, D=1024, H=16, AD=64,
// K_REL=16 (33 rel embeddings), EPS=1e-8.
// ---------------------------------------------------------------------------
typedef __attribute__((ext_vector_type(16))) __bf16       v16bf;
typedef __attribute__((ext_vector_type(8)))  float        v8f;
typedef __attribute__((ext_vector_type(4)))  unsigned int v4u;
typedef __attribute__((ext_vector_type(4)))  unsigned int u32x4;
typedef __attribute__((ext_vector_type(8)))  int          i32x8;
typedef __attribute__((ext_vector_type(4)))  int          i32x4;

union F8      { v8f  v; float f[8]; };
union BF16x16 { v16bf v; v4u u[2]; unsigned short s[16]; };

#if defined(__has_builtin)
#if __has_builtin(__builtin_amdgcn_tensor_load_to_lds) && \
    __has_builtin(__builtin_amdgcn_s_wait_tensorcnt)
#define USE_TDM 1
#endif
#endif

__device__ __forceinline__ __bf16 f32_to_bf16(float f) {
    union { float f; unsigned u; } in; in.f = f;
    unsigned u = in.u + 0x7FFFu + ((in.u >> 16) & 1u);   // round-to-nearest-even
    union { unsigned short s; __bf16 b; } out;
    out.s = (unsigned short)(u >> 16);
    return out.b;
}
__device__ __forceinline__ float bf16_to_f32(__bf16 b) {
    union { __bf16 b; unsigned short s; } in; in.b = b;
    union { unsigned u; float f; } out; out.u = ((unsigned)in.s) << 16;
    return out.f;
}

// A-matrix fragment (16x32 bf16, ISA 7.12.2): lane L holds row L&15;
// lane-half h holds K = h*8+0..7 (VGPR0-3) and 16+h*8+0..7 (VGPR4-7).
__device__ __forceinline__ v16bf load_a16(const __bf16* base, int ld) {
    const int lane = threadIdx.x & 31;
    const int r = lane & 15, h = lane >> 4;
    const __bf16* p = base + (size_t)r * ld + h * 8;
    BF16x16 t;
    t.u[0] = *(const v4u*)(p);        // cols h*8   .. h*8+7
    t.u[1] = *(const v4u*)(p + 16);   // cols 16+h*8.. 16+h*8+7
    return t.v;
}
// B-matrix fragment (32x16 bf16): lane L holds column L&15;
// lane-half h holds K = h*8+0..7 and 16+h*8+0..7 (matches A).
__device__ __forceinline__ v16bf load_b16(const __bf16* base, int ld) {
    const int lane = threadIdx.x & 31;
    const int c = lane & 15, h = lane >> 4;
    const __bf16* p = base + (size_t)c * ld + h * 8;
    BF16x16 t;
    t.u[0] = *(const v4u*)(p);
    t.u[1] = *(const v4u*)(p + 16);
    return t.v;
}

__device__ __forceinline__ v8f wmma_bf16(v16bf a, v16bf b, v8f c) {
    return __builtin_amdgcn_wmma_f32_16x16x32_bf16(
        /*neg_a=*/false, a, /*neg_b=*/false, b,
        /*c_mod=*/(short)0, c, /*reuse_a=*/false, /*reuse_b=*/false);
}

#ifdef USE_TDM
// Issue one 2D TDM tile load: global (tile_d1 rows x tile_d0 elems, bf16,
// row stride stride_elems) -> LDS at lds_off, rows packed contiguously.
// D# layout per CDNA5 ISA ch.8 (group0: count/lds/global/type, group1: dims).
__device__ __forceinline__ void tdm_load_2d(unsigned lds_off, const void* gaddr,
                                            unsigned tile_d0, unsigned tile_d1,
                                            unsigned stride_elems)
{
    unsigned long long ga = (unsigned long long)(uintptr_t)gaddr;
    u32x4 g0;
    g0[0] = 1u;                                            // count=1, user desc
    g0[1] = lds_off;                                       // lds_addr (bytes)
    g0[2] = (unsigned)(ga & 0xFFFFFFFFu);                  // global_addr[31:0]
    g0[3] = (unsigned)((ga >> 32) & 0x01FFFFFFu) | (2u << 30); // addr[56:32], type=2
    i32x8 g1;
    g1[0] = (int)(1u << 16);                               // data_size=1 (2 bytes)
    g1[1] = (int)((tile_d0 & 0xFFFFu) << 16);              // tensor_dim0[15:0]
    g1[2] = (int)((tile_d1 & 0xFFFFu) << 16);              // tensor_dim1[15:0]
    g1[3] = (int)((tile_d0 & 0xFFFFu) << 16);              // tile_dim0
    g1[4] = (int)(tile_d1 & 0xFFFFu);                      // tile_dim1 (tile_dim2=0)
    g1[5] = (int)stride_elems;                             // tensor_dim0_stride lo32
    g1[6] = 0;                                             // stride hi / dim1_stride lo
    g1[7] = 0;
    i32x4 z; z[0] = 0; z[1] = 0; z[2] = 0; z[3] = 0;
#if __clang_major__ >= 23
    i32x8 z8; z8[0]=0; z8[1]=0; z8[2]=0; z8[3]=0; z8[4]=0; z8[5]=0; z8[6]=0; z8[7]=0;
    __builtin_amdgcn_tensor_load_to_lds(g0, g1, z, z, z8, 0);
#else
    __builtin_amdgcn_tensor_load_to_lds(g0, g1, z, z, 0);
#endif
}
#endif

// ---------------------------------------------------------------------------
// fp32 -> bf16 conversion (inputs arrive as fp32)
// ---------------------------------------------------------------------------
__global__ void cvt_kernel(const float* __restrict__ src, __bf16* __restrict__ dst, int n) {
    int i = blockIdx.x * 256 + threadIdx.x;
    if (i < n) dst[i] = f32_to_bf16(src[i]);
}

// ---------------------------------------------------------------------------
// QKV GEMM: qkv[4096,3072] = iQ[4096,1024] @ Wa^T + ba
// Epilogue scatters into q [BH,S,AD] (pre-scaled by 1/8), k [BH,S,AD],
// and V TRANSPOSED vt [BH,AD,S]. B fragments loaded as a group of 4 so the
// scheduler keeps 4 b128 loads in flight per WMMA group.
// ---------------------------------------------------------------------------
__global__ void gemm_qkv_kernel(const __bf16* __restrict__ A,
                                const __bf16* __restrict__ W,
                                const float*  __restrict__ ba,
                                __bf16* __restrict__ qb,
                                __bf16* __restrict__ kb,
                                __bf16* __restrict__ vtb)
{
    const int lane = threadIdx.x & 31;
    const int w    = threadIdx.x >> 5;
    const int n0   = blockIdx.x * 64;
    const int m0   = blockIdx.y * 64 + w * 16;
    const int N = lane & 15, hh = lane >> 4;

    F8 acc[4];
    #pragma unroll
    for (int t = 0; t < 4; ++t) {
        const float bias = ba[n0 + t * 16 + N];
        #pragma unroll
        for (int r = 0; r < 8; ++r) acc[t].f[r] = bias;
    }
    for (int kk = 0; kk < 1024; kk += 32) {
        v16bf a  = load_a16(A + (size_t)m0 * 1024 + kk, 1024);
        v16bf b0 = load_b16(W + (size_t)(n0 +  0) * 1024 + kk, 1024);
        v16bf b1 = load_b16(W + (size_t)(n0 + 16) * 1024 + kk, 1024);
        v16bf b2 = load_b16(W + (size_t)(n0 + 32) * 1024 + kk, 1024);
        v16bf b3 = load_b16(W + (size_t)(n0 + 48) * 1024 + kk, 1024);
        acc[0].v = wmma_bf16(a, b0, acc[0].v);
        acc[1].v = wmma_bf16(a, b1, acc[1].v);
        acc[2].v = wmma_bf16(a, b2, acc[2].v);
        acc[3].v = wmma_bf16(a, b3, acc[3].v);
    }
    #pragma unroll
    for (int t = 0; t < 4; ++t) {
        const int n    = n0 + t * 16 + N;
        const int comp = n >> 10;            // 0=q 1=k 2=v
        const int nd   = n & 1023;
        const int h    = nd >> 6;
        const int d    = nd & 63;
        #pragma unroll
        for (int r = 0; r < 8; ++r) {
            const int m = m0 + r + 8 * hh;
            const int b = m >> 10, s = m & 1023;
            const int bh = b * 16 + h;
            const float v = acc[t].f[r];
            if (comp == 0)       qb [((size_t)bh * 1024 + s) * 64 + d] = f32_to_bf16(v * 0.125f);
            else if (comp == 1)  kb [((size_t)bh * 1024 + s) * 64 + d] = f32_to_bf16(v);
            else                 vtb[((size_t)bh * 64 + d) * 1024 + s] = f32_to_bf16(v);
        }
    }
}

// ---------------------------------------------------------------------------
// qrel[BH*S, 33] = q . rel_emb^T  (33 distinct rel embeddings; q pre-scaled)
// ---------------------------------------------------------------------------
__global__ void qrel_kernel(const __bf16* __restrict__ qb,
                            const float*  __restrict__ rel,
                            float* __restrict__ qr)
{
    int idx = blockIdx.x * 256 + threadIdx.x;
    if (idx >= 65536 * 33) return;
    int row = idx / 33, p = idx - row * 33;
    const __bf16* q = qb + (size_t)row * 64;
    const float*  r = rel + p * 64;
    float s = 0.f;
    #pragma unroll 8
    for (int d = 0; d < 64; ++d) s += bf16_to_f32(q[d]) * r[d];
    qr[idx] = s;
}

// ---------------------------------------------------------------------------
// Fused streaming attention. SparseNormer is linear in t=relu(x+b)^2, so one
// pass accumulates U = sum_k t*v and denom = sum_k t (denom via WMMA x ones);
// no online rescaling. One block = one (b,h) x 64 q-rows, 4 waves x 16 rows.
// K/V tiles (32 keys) are staged into double-buffered LDS by wave 0 via the
// Tensor Data Mover (tensor_load_to_lds + s_wait_tensorcnt) when available,
// falling back to direct global b128 fragment loads otherwise.
// ---------------------------------------------------------------------------
__global__ void attn_kernel(const __bf16* __restrict__ qb,
                            const __bf16* __restrict__ kb,
                            const __bf16* __restrict__ vtb,
                            const float*  __restrict__ qr,
                            const float*  __restrict__ sn_bias,
                            __bf16* __restrict__ ctxb)
{
    __shared__ __align__(16) __bf16 sh_t[4][16][32];   // per-wave t staging
    __shared__ float shq[4][16][34];                   // per-wave qrel tile
#ifdef USE_TDM
    __shared__ __align__(16) __bf16 ktile[2][32][64];  // [key][d]   8KB
    __shared__ __align__(16) __bf16 vtile[2][64][32];  // [d][key]   8KB
#endif

    const int lane  = threadIdx.x & 31;
    const int w     = threadIdx.x >> 5;
    const int qblk  = blockIdx.x;
    const int bh    = blockIdx.y;
    const int b     = bh >> 4, h = bh & 15;
    const int qrow0 = qblk * 64 + w * 16;
    const size_t rowbase = (size_t)bh * 1024;
    const int N = lane & 15, hh = lane >> 4;
    const float snb = sn_bias[0];

    for (int idx = lane; idx < 16 * 33; idx += 32) {   // wave-private stage
        int rr = idx / 33, pp = idx - rr * 33;
        shq[w][rr][pp] = qr[(rowbase + qrow0 + rr) * 33 + pp];
    }

    const v16bf aq0 = load_a16(qb + (rowbase + qrow0) * 64, 64);
    const v16bf aq1 = load_a16(qb + (rowbase + qrow0) * 64 + 32, 64);

    BF16x16 ones;
    #pragma unroll
    for (int i = 0; i < 16; ++i) ones.s[i] = 0x3F80;   // bf16 1.0

    F8 accC[4], accD;
    #pragma unroll
    for (int t = 0; t < 4; ++t)
        #pragma unroll
        for (int r = 0; r < 8; ++r) accC[t].f[r] = 0.f;
    #pragma unroll
    for (int r = 0; r < 8; ++r) accD.f[r] = 0.f;

#ifdef USE_TDM
    if (w == 0) {   // prologue: DMA tile 0 (K: 32x64 rows-contig, V: 64x32)
        tdm_load_2d((unsigned)(uintptr_t)&ktile[0][0][0],
                    kb + (rowbase + 0) * 64, 64, 32, 64);
        tdm_load_2d((unsigned)(uintptr_t)&vtile[0][0][0],
                    vtb + (size_t)bh * 64 * 1024 + 0, 32, 64, 1024);
    }
#endif

    for (int kk = 0; kk < 1024; kk += 32) {
#ifdef USE_TDM
        const int cur = (kk >> 5) & 1;
        if (w == 0) {
            if (kk + 32 < 1024) {   // prefetch next tile into other buffer
                tdm_load_2d((unsigned)(uintptr_t)&ktile[cur ^ 1][0][0],
                            kb + (rowbase + kk + 32) * 64, 64, 32, 64);
                tdm_load_2d((unsigned)(uintptr_t)&vtile[cur ^ 1][0][0],
                            vtb + (size_t)bh * 64 * 1024 + kk + 32, 32, 64, 1024);
                __builtin_amdgcn_s_wait_tensorcnt(2);  // current pair done
            } else {
                __builtin_amdgcn_s_wait_tensorcnt(0);
            }
        }
        __syncthreads();            // LDS K/V tile visible to all waves
        v16bf bk0a = load_b16(&ktile[cur][ 0][0], 64);
        v16bf bk0b = load_b16(&ktile[cur][ 0][32], 64);
        v16bf bk1a = load_b16(&ktile[cur][16][0], 64);
        v16bf bk1b = load_b16(&ktile[cur][16][32], 64);
#else
        v16bf bk0a = load_b16(kb + (rowbase + kk) * 64, 64);
        v16bf bk0b = load_b16(kb + (rowbase + kk) * 64 + 32, 64);
        v16bf bk1a = load_b16(kb + (rowbase + kk + 16) * 64, 64);
        v16bf bk1b = load_b16(kb + (rowbase + kk + 16) * 64 + 32, 64);
        if (kk + 32 < 1024) {  // gfx1250 global_prefetch_b8 for next K/V tiles
            __builtin_prefetch(kb + (rowbase + kk + 32 + lane) * 64, 0, 1);
            __builtin_prefetch(vtb + ((size_t)bh * 64 + lane) * 1024 + kk + 32, 0, 1);
            __builtin_prefetch(vtb + ((size_t)bh * 64 + 32 + lane) * 1024 + kk + 32, 0, 1);
        }
#endif
        // scores for two 16-key subtiles, K=64 split into two K=32 WMMAs
        F8 s0, s1;
        #pragma unroll
        for (int r = 0; r < 8; ++r) { s0.f[r] = 0.f; s1.f[r] = 0.f; }
        s0.v = wmma_bf16(aq0, bk0a, s0.v);
        s0.v = wmma_bf16(aq1, bk0b, s0.v);
        s1.v = wmma_bf16(aq0, bk1a, s1.v);
        s1.v = wmma_bf16(aq1, bk1b, s1.v);

        // x = qk + qrel[clip(j-i)] + bias;  t = relu(x)^2 ;  D-layout -> LDS
        #pragma unroll
        for (int r = 0; r < 8; ++r) {
            const int M  = r + 8 * hh;
            const int qi = qrow0 + M;
            int dlt = (kk + N) - qi;
            dlt = dlt < -16 ? -16 : (dlt > 16 ? 16 : dlt);
            float x = s0.f[r] + shq[w][M][dlt + 16] + snb;
            x = x > 0.f ? x * x : 0.f;
            sh_t[w][M][N] = f32_to_bf16(x);
            dlt = (kk + 16 + N) - qi;
            dlt = dlt < -16 ? -16 : (dlt > 16 ? 16 : dlt);
            float y = s1.f[r] + shq[w][M][dlt + 16] + snb;
            y = y > 0.f ? y * y : 0.f;
            sh_t[w][M][N + 16] = f32_to_bf16(y);
        }

        // re-load t in A-layout (wave-private LDS; DS in-order per wave)
        const v16bf at = load_a16(&sh_t[w][0][0], 32);
        accD.v = wmma_bf16(at, ones.v, accD.v);            // row sums
        #pragma unroll
        for (int tn = 0; tn < 4; ++tn) {
#ifdef USE_TDM
            v16bf bv = load_b16(&vtile[cur][tn * 16][0], 32);
#else
            v16bf bv = load_b16(vtb + ((size_t)bh * 64 + tn * 16) * 1024 + kk, 1024);
#endif
            accC[tn].v = wmma_bf16(at, bv, accC[tn].v);    // U += t @ V
        }
#ifdef USE_TDM
        __syncthreads();            // all waves done with buf[cur] before reuse
#endif
    }

    // ctx = U / (denom + eps); write token-major [B,S,H*AD] for the out-GEMM
    #pragma unroll
    for (int r = 0; r < 8; ++r) {
        const int M = r + 8 * hh;
        const int s = qrow0 + M;
        const float inv = 1.0f / (accD.f[r] + 1e-8f);
        #pragma unroll
        for (int tn = 0; tn < 4; ++tn)
            ctxb[((size_t)b * 1024 + s) * 1024 + h * 64 + tn * 16 + N] =
                f32_to_bf16(accC[tn].f[r] * inv);
    }
}

// ---------------------------------------------------------------------------
// Output GEMM: out[4096,1024] = ctx @ Wo^T + bo  (fp32 output)
// ---------------------------------------------------------------------------
__global__ void gemm_out_kernel(const __bf16* __restrict__ A,
                                const __bf16* __restrict__ W,
                                const float*  __restrict__ bo,
                                float* __restrict__ out)
{
    const int lane = threadIdx.x & 31;
    const int w    = threadIdx.x >> 5;
    const int n0   = blockIdx.x * 64;
    const int m0   = blockIdx.y * 64 + w * 16;
    const int N = lane & 15, hh = lane >> 4;

    F8 acc[4];
    #pragma unroll
    for (int t = 0; t < 4; ++t) {
        const float bias = bo[n0 + t * 16 + N];
        #pragma unroll
        for (int r = 0; r < 8; ++r) acc[t].f[r] = bias;
    }
    for (int kk = 0; kk < 1024; kk += 32) {
        v16bf a  = load_a16(A + (size_t)m0 * 1024 + kk, 1024);
        v16bf b0 = load_b16(W + (size_t)(n0 +  0) * 1024 + kk, 1024);
        v16bf b1 = load_b16(W + (size_t)(n0 + 16) * 1024 + kk, 1024);
        v16bf b2 = load_b16(W + (size_t)(n0 + 32) * 1024 + kk, 1024);
        v16bf b3 = load_b16(W + (size_t)(n0 + 48) * 1024 + kk, 1024);
        acc[0].v = wmma_bf16(a, b0, acc[0].v);
        acc[1].v = wmma_bf16(a, b1, acc[1].v);
        acc[2].v = wmma_bf16(a, b2, acc[2].v);
        acc[3].v = wmma_bf16(a, b3, acc[3].v);
    }
    #pragma unroll
    for (int t = 0; t < 4; ++t)
        #pragma unroll
        for (int r = 0; r < 8; ++r)
            out[(size_t)(m0 + r + 8 * hh) * 1024 + n0 + t * 16 + N] = acc[t].f[r];
}

// ---------------------------------------------------------------------------
extern "C" void kernel_launch(void* const* d_in, const int* in_sizes, int n_in,
                              void* d_out, int out_size, void* d_ws, size_t ws_size,
                              hipStream_t stream) {
    (void)in_sizes; (void)n_in; (void)out_size; (void)ws_size;
    const float* iQ  = (const float*)d_in[0];
    const float* Wa  = (const float*)d_in[1];
    const float* ba  = (const float*)d_in[2];
    const float* rel = (const float*)d_in[3];
    const float* snb = (const float*)d_in[4];
    const float* Wo  = (const float*)d_in[5];
    const float* bo  = (const float*)d_in[6];

    char* ws = (char*)d_ws;
    size_t off = 0;
    auto take = [&](size_t bytes) -> char* {
        char* p = ws + off;
        off += (bytes + 255) & ~(size_t)255;
        return p;
    };
    __bf16* iQb  = (__bf16*)take((size_t)4194304 * 2);  // [4096,1024]
    __bf16* Wab  = (__bf16*)take((size_t)3145728 * 2);  // [3072,1024]
    __bf16* Wob  = (__bf16*)take((size_t)1048576 * 2);  // [1024,1024]
    __bf16* qb   = (__bf16*)take((size_t)4194304 * 2);  // [BH,S,AD]
    __bf16* kb   = (__bf16*)take((size_t)4194304 * 2);  // [BH,S,AD]
    __bf16* vtb  = (__bf16*)take((size_t)4194304 * 2);  // [BH,AD,S]
    float*  qrw  = (float*) take((size_t)2162688 * 4);  // [BH*S,33]
    __bf16* ctxb = (__bf16*)take((size_t)4194304 * 2);  // [B,S,D]

    cvt_kernel<<<4194304 / 256, 256, 0, stream>>>(iQ, iQb, 4194304);
    cvt_kernel<<<3145728 / 256, 256, 0, stream>>>(Wa, Wab, 3145728);
    cvt_kernel<<<1048576 / 256, 256, 0, stream>>>(Wo, Wob, 1048576);

    gemm_qkv_kernel<<<dim3(48, 64), 128, 0, stream>>>(iQb, Wab, ba, qb, kb, vtb);
    qrel_kernel<<<(2162688 + 255) / 256, 256, 0, stream>>>(qb, rel, qrw);
    attn_kernel<<<dim3(16, 64), 128, 0, stream>>>(qb, kb, vtb, qrw, snb, ctxb);
    gemm_out_kernel<<<dim3(16, 64), 128, 0, stream>>>(ctxb, Wob, bo, (float*)d_out);
}
